// MultiAttention_22926535426057
// MI455X (gfx1250) — compile-verified, matrix operand order checked by reading, not run on hardware
//
#include <hip/hip_runtime.h>
#include <hip/hip_bf16.h>
#include <math.h>

#define EMB    1024
#define HEADS  16
#define HEAD_D 128
#define NB     2
#define SEQ    2048
#define EOUT   2048   // EMB*EXPAND

typedef __attribute__((ext_vector_type(16))) __bf16 v16bf;
typedef __attribute__((ext_vector_type(8)))  float  v8f;

// CDNA5 16-bit A-matrix 16x32 layout (CDNA5 ISA §7.12.2): lane L holds row
// L&15; its 16 bf16 elements are two contiguous 16B chunks of the 32-element
// K-slab: [8g .. 8g+7] and [16+8g .. 16+8g+7], g = L>>4. B-fragments load the
// same pattern from the transposed operand's rows. Load as uint4 pairs.
union FragU {
  uint4 q[2];
  v16bf v;
};

union Pack8 {            // 8 f32 -> 8 bf16 -> one 16B LDS store
  __bf16 h[8];
  uint4  q;
};

// ---------------------------------------------------------------------------
// Mask bit-pack: bits[w] = ballot over 32 consecutive mask words.
// ---------------------------------------------------------------------------
__global__ void __launch_bounds__(256)
pack_mask(const int* __restrict__ mask, unsigned* __restrict__ bits)
{
  int gid = blockIdx.x * 256 + threadIdx.x;          // 0 .. S*S-1
  unsigned long long b = __ballot(mask[gid] != 0);   // wave32: low 32 bits
  if ((threadIdx.x & 31) == 0) bits[gid >> 5] = (unsigned)b;
}

// ---------------------------------------------------------------------------
// Projection: dst = relu(X @ W^T + b) cast to bf16, LDS-tiled.
//   Block = 256 thr (8 waves), block tile 64M x 128N, K-step 32.
//   Stages LA[64][32] + LB[128][32] bf16 (12KB); each f32 is loaded and
//   converted once per block instead of once per wave.
//   vtrans==0: dst[((n*H+h)*SEQ + s)*HEAD_D + d]   (Q, K)
//   vtrans==1: dst[((n*H+h)*HEAD_D + d)*SEQ + s]   (V, transposed)
// ---------------------------------------------------------------------------
__global__ void __launch_bounds__(256)
proj_relu_bf16(const float* __restrict__ X, const float* __restrict__ W,
               const float* __restrict__ b, __bf16* __restrict__ dst, int vtrans)
{
  __shared__ __attribute__((aligned(16))) __bf16 LA[64][32];    // 4KB
  __shared__ __attribute__((aligned(16))) __bf16 LB[128][32];   // 8KB

  const int tid  = threadIdx.x;
  const int lane = tid & 31;
  const int wave = tid >> 5;
  const int g  = lane >> 4;
  const int lr = lane & 15;
  const int wm = wave >> 1;          // 0..3: 16-row strip
  const int wn = wave & 1;           // 0..1: 64-col strip

  const int m0 = blockIdx.x * 64;
  const int n0 = blockIdx.y * 128;

  v8f acc[4];
  #pragma unroll
  for (int t = 0; t < 4; ++t) { v8f z = {}; acc[t] = z; }

  for (int k0 = 0; k0 < EMB; k0 += 32) {
    __syncthreads();
    // ---- stage A: 64x32 f32 -> bf16 ; thread: row=tid>>2, 8 k's ----
    {
      int row = tid >> 2, ko = (tid & 3) * 8;
      const float* src = X + (size_t)(m0 + row) * EMB + k0 + ko;
      float4 f0 = *(const float4*)src;
      float4 f1 = *(const float4*)(src + 4);
      Pack8 u;
      u.h[0] = (__bf16)f0.x; u.h[1] = (__bf16)f0.y;
      u.h[2] = (__bf16)f0.z; u.h[3] = (__bf16)f0.w;
      u.h[4] = (__bf16)f1.x; u.h[5] = (__bf16)f1.y;
      u.h[6] = (__bf16)f1.z; u.h[7] = (__bf16)f1.w;
      *(uint4*)(&LA[row][ko]) = u.q;
    }
    // ---- stage B: 128x32 f32 -> bf16 ; thread: row=tid>>1, 16 k's ----
    {
      int row = tid >> 1, ko = (tid & 1) * 16;
      const float* src = W + (size_t)(n0 + row) * EMB + k0 + ko;
      float4 f0 = *(const float4*)src;
      float4 f1 = *(const float4*)(src + 4);
      float4 f2 = *(const float4*)(src + 8);
      float4 f3 = *(const float4*)(src + 12);
      Pack8 u0, u1;
      u0.h[0] = (__bf16)f0.x; u0.h[1] = (__bf16)f0.y;
      u0.h[2] = (__bf16)f0.z; u0.h[3] = (__bf16)f0.w;
      u0.h[4] = (__bf16)f1.x; u0.h[5] = (__bf16)f1.y;
      u0.h[6] = (__bf16)f1.z; u0.h[7] = (__bf16)f1.w;
      u1.h[0] = (__bf16)f2.x; u1.h[1] = (__bf16)f2.y;
      u1.h[2] = (__bf16)f2.z; u1.h[3] = (__bf16)f2.w;
      u1.h[4] = (__bf16)f3.x; u1.h[5] = (__bf16)f3.y;
      u1.h[6] = (__bf16)f3.z; u1.h[7] = (__bf16)f3.w;
      *(uint4*)(&LB[row][ko])     = u0.q;
      *(uint4*)(&LB[row][ko + 8]) = u1.q;
    }
    __syncthreads();

    // ---- wave tile: 16M x 64N, operands from LDS ----
    FragU a;
    const __bf16* ar = &LA[wm * 16 + lr][8 * g];
    a.q[0] = *(const uint4*)ar;
    a.q[1] = *(const uint4*)(ar + 16);
    #pragma unroll
    for (int t = 0; t < 4; ++t) {
      FragU bb;
      const __bf16* br = &LB[wn * 64 + t * 16 + lr][8 * g];
      bb.q[0] = *(const uint4*)br;
      bb.q[1] = *(const uint4*)(br + 16);
      acc[t] = __builtin_amdgcn_wmma_f32_16x16x32_bf16(
          false, a.v, false, bb.v, (short)0, acc[t], false, false);
    }
  }

  #pragma unroll
  for (int t = 0; t < 4; ++t) {
    #pragma unroll
    for (int v = 0; v < 8; ++v) {
      int orow = m0 + wm * 16 + v + 8 * g;            // n*SEQ + s
      int ocol = n0 + wn * 64 + t * 16 + lr;          // h*128 + d
      float val = fmaxf(acc[t][v] + b[ocol], 0.0f);
      int n = orow >> 11, s = orow & (SEQ - 1);
      int h = ocol >> 7,  d = ocol & (HEAD_D - 1);
      size_t idx = vtrans
          ? ((size_t)(n * HEADS + h) * HEAD_D + d) * SEQ + s
          : ((size_t)(n * HEADS + h) * SEQ + s) * HEAD_D + d;
      dst[idx] = (__bf16)val;
    }
  }
}

// ---------------------------------------------------------------------------
// Flash attention. Block = 256 thr (8 waves) sharing one (n,h); wave w owns
// q-tile (blockIdx&15)*8 + w. Per 32-key tile the block stages K (8KB) and
// V^T (8KB) into LDS once for all 8 waves; hot-loop operands are ds_load_b128.
// Mask comes from the bit-packed table: one u32 per score row per tile.
// ---------------------------------------------------------------------------
__global__ void __launch_bounds__(256)
flash_attn_bf16(const __bf16* __restrict__ Qb, const __bf16* __restrict__ Kb,
                const __bf16* __restrict__ Vt, const unsigned* __restrict__ MB,
                float* __restrict__ out)
{
  __shared__ __attribute__((aligned(16))) __bf16 LK[32][128];     // 8KB
  __shared__ __attribute__((aligned(16))) __bf16 LV[128][32];     // 8KB
  __shared__ __attribute__((aligned(16))) __bf16 pst[8][16][32];  // 8KB

  const int tid  = threadIdx.x;
  const int lane = tid & 31;
  const int wave = tid >> 5;
  const int g  = lane >> 4;
  const int lr = lane & 15;

  const int nh    = blockIdx.x >> 4;               // 0..31
  const int n     = nh >> 4;
  const int h     = nh & (HEADS - 1);
  const int qtile = (blockIdx.x & 15) * 8 + wave;  // 0..127

  const __bf16* Qh = Qb + (size_t)(n * HEADS + h) * SEQ * HEAD_D;
  const __bf16* Kh = Kb + (size_t)(n * HEADS + h) * SEQ * HEAD_D;
  const __bf16* Vh = Vt + (size_t)(n * HEADS + h) * HEAD_D * SEQ;

  // Preload Q fragments (16 q-rows x 128 d) as uint4 pairs.
  v16bf qa[4];
  const __bf16* qp = Qh + (size_t)(qtile * 16 + lr) * HEAD_D;
  #pragma unroll
  for (int kk = 0; kk < 4; ++kk) {
    FragU u;
    u.q[0] = *(const uint4*)(qp + kk * 32 + 8 * g);
    u.q[1] = *(const uint4*)(qp + kk * 32 + 16 + 8 * g);
    qa[kk] = u.v;
  }

  float m[8], l[8];
  v8f o[8];
  #pragma unroll
  for (int v = 0; v < 8; ++v) { m[v] = -3.0e38f; l[v] = 0.0f; }
  #pragma unroll
  for (int t = 0; t < 8; ++t) { v8f z = {}; o[t] = z; }

  const float scale = 0.08838834764831845f;  // 1/sqrt(128)

  for (int j0 = 0; j0 < SEQ; j0 += 32) {
    // ---- cooperative stage of K/V tiles (512+512 16B chunks / 256 thr) ----
    __syncthreads();
    {
      const uint4* gk = (const uint4*)(Kh + (size_t)j0 * HEAD_D);  // 8KB contig
      uint4* lk = (uint4*)(&LK[0][0]);
      lk[tid]       = gk[tid];
      lk[tid + 256] = gk[tid + 256];

      const __bf16* gv = Vh + j0;              // row stride SEQ, 64B per row
      uint4* lv = (uint4*)(&LV[0][0]);
      #pragma unroll
      for (int c = 0; c < 2; ++c) {
        int ch = tid + 256 * c;                // d = ch>>2, 16B piece = ch&3
        lv[ch] = *(const uint4*)(gv + (size_t)(ch >> 2) * SEQ + (ch & 3) * 8);
      }
    }
    if (j0 + 32 < SEQ)
      __builtin_prefetch(Kh + (size_t)(j0 + 32) * HEAD_D + tid * 16, 0, 1);
    __syncthreads();

    // ---- scores: 16 q x 32 k, contraction over D=128, operands from LDS ----
    v8f s0 = {}, s1 = {};
    #pragma unroll
    for (int kk = 0; kk < 4; ++kk) {
      FragU b0, b1;
      const __bf16* r0 = &LK[lr][kk * 32 + 8 * g];
      const __bf16* r1 = &LK[16 + lr][kk * 32 + 8 * g];
      b0.q[0] = *(const uint4*)r0; b0.q[1] = *(const uint4*)(r0 + 16);
      b1.q[0] = *(const uint4*)r1; b1.q[1] = *(const uint4*)(r1 + 16);
      s0 = __builtin_amdgcn_wmma_f32_16x16x32_bf16(false, qa[kk], false, b0.v,
                                                   (short)0, s0, false, false);
      s1 = __builtin_amdgcn_wmma_f32_16x16x32_bf16(false, qa[kk], false, b1.v,
                                                   (short)0, s1, false, false);
    }

    // ---- bit-mask + scale + online softmax (row = v + 8*g, col = lr) ----
    #pragma unroll
    for (int v = 0; v < 8; ++v) {
      int q = qtile * 16 + v + 8 * g;
      unsigned mb = MB[(size_t)q * (SEQ / 32) + (j0 >> 5)];
      float x0 = s0[v] * scale;
      float x1 = s1[v] * scale;
      if (!((mb >> lr) & 1u))        x0 = -1.0e7f;
      if (!((mb >> (16 + lr)) & 1u)) x1 = -1.0e7f;

      float mx = fmaxf(x0, x1);
      mx = fmaxf(mx, __shfl_xor(mx, 1));
      mx = fmaxf(mx, __shfl_xor(mx, 2));
      mx = fmaxf(mx, __shfl_xor(mx, 4));
      mx = fmaxf(mx, __shfl_xor(mx, 8));

      float mn    = fmaxf(m[v], mx);
      float alpha = __expf(m[v] - mn);
      float p0    = __expf(x0 - mn);
      float p1    = __expf(x1 - mn);

      float rs = p0 + p1;
      rs += __shfl_xor(rs, 1);
      rs += __shfl_xor(rs, 2);
      rs += __shfl_xor(rs, 4);
      rs += __shfl_xor(rs, 8);

      l[v] = l[v] * alpha + rs;
      m[v] = mn;
      #pragma unroll
      for (int t = 0; t < 8; ++t) o[t][v] *= alpha;

      pst[wave][v + 8 * g][lr]      = (__bf16)p0;
      pst[wave][v + 8 * g][16 + lr] = (__bf16)p1;
    }

    // ---- re-layout P (C-format, per-wave LDS) into an A-fragment ----
    FragU pa;
    const __bf16* pr = &pst[wave][lr][8 * g];
    pa.q[0] = *(const uint4*)pr;
    pa.q[1] = *(const uint4*)(pr + 16);

    // ---- o += P @ V : B-frags from LDS V^T rows (d-major, s-contiguous) ----
    #pragma unroll
    for (int t = 0; t < 8; ++t) {
      FragU vb;
      const __bf16* vr = &LV[t * 16 + lr][8 * g];
      vb.q[0] = *(const uint4*)vr;
      vb.q[1] = *(const uint4*)(vr + 16);
      o[t] = __builtin_amdgcn_wmma_f32_16x16x32_bf16(false, pa.v, false, vb.v,
                                                     (short)0, o[t], false, false);
    }
  }

  // ---- epilogue: y = o / l, scatter to (n, s, h*128 + d) ----
  #pragma unroll
  for (int t = 0; t < 8; ++t) {
    #pragma unroll
    for (int v = 0; v < 8; ++v) {
      int q = qtile * 16 + v + 8 * g;
      int c = h * HEAD_D + t * 16 + lr;
      out[((size_t)n * SEQ + q) * EOUT + c] = o[t][v] / l[v];
    }
  }
}

extern "C" void kernel_launch(void* const* d_in, const int* in_sizes, int n_in,
                              void* d_out, int out_size, void* d_ws, size_t ws_size,
                              hipStream_t stream)
{
  const float* query = (const float*)d_in[0];
  const float* key   = (const float*)d_in[1];
  const float* value = (const float*)d_in[2];
  const int*   mask  = (const int*)d_in[3];
  const float* Wq = (const float*)d_in[4];
  const float* bq = (const float*)d_in[5];
  const float* Wk = (const float*)d_in[6];
  const float* bk = (const float*)d_in[7];
  const float* Wv = (const float*)d_in[8];
  const float* bv = (const float*)d_in[9];
  float* out = (float*)d_out;

  const size_t elems = (size_t)NB * SEQ * EOUT;   // 8M bf16 each
  __bf16* Qb = (__bf16*)d_ws;
  __bf16* Kb = Qb + elems;
  __bf16* Vt = Kb + elems;
  unsigned* maskbits = (unsigned*)(Vt + elems);   // 512KB

  pack_mask<<<dim3((SEQ * SEQ) / 256, 1, 1), dim3(256, 1, 1), 0, stream>>>(
      mask, maskbits);

  dim3 pgrid(64, 16, 1), pblk(256, 1, 1);
  proj_relu_bf16<<<pgrid, pblk, 0, stream>>>(query, Wq, bq, Qb, 0);
  proj_relu_bf16<<<pgrid, pblk, 0, stream>>>(key,   Wk, bk, Kb, 0);
  proj_relu_bf16<<<pgrid, pblk, 0, stream>>>(value, Wv, bv, Vt, 1);

  flash_attn_bf16<<<dim3(512, 1, 1), dim3(256, 1, 1), 0, stream>>>(
      Qb, Kb, Vt, maskbits, out);
}
